// SRNO_net_46772193853703
// MI455X (gfx1250) — matrix operand order repaired and checked by reading, not compiled, not run on hardware
//
#include <hip/hip_runtime.h>
#include <hip/hip_bf16.h>
#include <math.h>

// ---------------------------------------------------------------------------
// SRNO forward for MI455X (gfx1250, wave32, WMMA).
// All dense math runs through v_wmma_f32_16x16x32_f16 (f16 in, f32 acc).
// GEMM tiles are moved global->LDS with CDNA5 async copies
// (global_load_async_to_lds_b128, ASYNCcnt) and double-buffered.
// Weights are kept in transposed [N][Kpad] layout so every WMMA fragment is
// two contiguous 16B runs in LDS (vectorized ds_load_b128, no dscnt stalls).
//
// Input pointer order: setup_inputs() dict insertion order, params flattened
// recursively in insertion order (see round-0 comment; indices 0..40).
// ---------------------------------------------------------------------------

typedef __attribute__((ext_vector_type(16))) _Float16 v16h;
typedef __attribute__((ext_vector_type(8)))  float    v8f;

#define BM 128
#define BN 64
#define BK 32
#define LDA 40            // halves; 80B row stride (16B multiple)
#define LDBT 40           // halves; 80B row stride (16B multiple)
#define SLICE (BM * LDA + BN * LDBT)   // halves per double-buffer slice

#define ACT_NONE 0
#define ACT_RELU 1
#define ACT_GELU 2

// CDNA5 async copy: per-lane 16B global -> LDS, tracked with ASYNCcnt.
__device__ __forceinline__ void async_copy_b128(unsigned lds_off, const void* gptr)
{
    asm volatile("global_load_async_to_lds_b128 %0, %1, off"
                 :: "v"(lds_off), "v"(gptr) : "memory");
}
__device__ __forceinline__ void wait_async0()
{
    asm volatile("s_wait_asynccnt 0" ::: "memory");
}

// ---------------------------------------------------------------------------
// D[M,N](f16) = act(A[M,K]h * Bt[N,K]h^T + bias) (+ res), f32 accumulate.
// M%128==0, N%64==0, K%32==0 (padded). 8 waves; wave owns 32x32 = 2x2 frags.
// ---------------------------------------------------------------------------
__global__ __launch_bounds__(256, 2) void gemm_f16_wmma(
    const _Float16* __restrict__ A, const _Float16* __restrict__ Bt,
    const float* __restrict__ bias, const _Float16* __restrict__ res,
    _Float16* __restrict__ D, int M, int N, int K, int act)
{
    // single shared object -> LDS offset 0 assumption for async-copy asm
    __shared__ alignas(16) _Float16 smem[2 * SLICE];

    const int tid  = threadIdx.x;
    const int lane = tid & 31;
    const int wave = tid >> 5;
    const int wm   = wave & 3;    // 32-row strip
    const int wn   = wave >> 2;   // 32-col strip
    const int m_blk = blockIdx.x * BM;
    const int n_blk = blockIdx.y * BN;

    // tile-load assignments (16B per async op)
    const int ar  = tid >> 1;          // 0..127 (A row)
    const int ac  = (tid & 1) << 4;    // 0/16 halves
    const int bn  = tid >> 2;          // 0..63 (B row = output channel)
    const int bc  = (tid & 3) << 3;    // 0..24 halves

    const _Float16* agBase = A  + (size_t)(m_blk + ar) * K + ac;
    const _Float16* bgBase = Bt + (size_t)(n_blk + bn) * K + bc;
    const unsigned ldsA0 = (unsigned)((ar * LDA + ac) * 2);
    const unsigned ldsB0 = (unsigned)((BM * LDA + bn * LDBT + bc) * 2);

    auto issue = [&](int k0, int buf) {
        const unsigned off = (unsigned)(buf * (SLICE * 2));
        async_copy_b128(off + ldsA0,      agBase + k0);
        async_copy_b128(off + ldsA0 + 16, agBase + k0 + 8);
        async_copy_b128(off + ldsB0,      bgBase + k0);
    };

    v8f acc[2][2] = {};
    issue(0, 0);
    int cur = 0;

    for (int k0 = 0; k0 < K; k0 += BK) {
        wait_async0();        // this wave's copies for buffer `cur` landed
        __syncthreads();      // everyone's copies landed; prev reads done
        if (k0 + BK < K) issue(k0 + BK, cur ^ 1);

        const _Float16* As = &smem[cur * SLICE];
        const _Float16* Bs = As + BM * LDA;

        const int r  = lane & 15;
        const int ka = (lane >> 4) << 3;   // A half-wave K offset (0/8)
        const int kb = (lane >> 4) << 3;   // B half-wave K offset (0/8)

        v16h af[2], bf[2];
        #pragma unroll
        for (int im = 0; im < 2; ++im) {
            const _Float16* ap = &As[(wm * 32 + im * 16 + r) * LDA + ka];
            #pragma unroll
            for (int i = 0; i < 8; ++i) {
                // A 16x32: VGPR0-3 -> K 0..7, VGPR4-7 -> K 16..23 (+ka)
                int k = (i < 4) ? (i << 1) : (16 + ((i - 4) << 1));
                af[im][2 * i]     = ap[k];
                af[im][2 * i + 1] = ap[k + 1];
            }
        }
        #pragma unroll
        for (int in = 0; in < 2; ++in) {
            const _Float16* bp = &Bs[(wn * 32 + in * 16 + r) * LDBT + kb];
            #pragma unroll
            for (int i = 0; i < 8; ++i) {
                // B 32x16: VGPR0-3 -> K 0..7, VGPR4-7 -> K 16..23 (+kb)
                int k = (i < 4) ? (i << 1) : (16 + ((i - 4) << 1));
                bf[in][2 * i]     = bp[k];
                bf[in][2 * i + 1] = bp[k + 1];
            }
        }
        #pragma unroll
        for (int im = 0; im < 2; ++im)
            #pragma unroll
            for (int in = 0; in < 2; ++in)
                acc[im][in] = __builtin_amdgcn_wmma_f32_16x16x32_f16(
                    false, af[im], false, bf[in], (short)0, acc[im][in],
                    false, false);
        cur ^= 1;
    }

    // epilogue: lane&15 = N col, lane>>4 selects M rows 0-7 / 8-15
    const int nl = lane & 15;
    const int mh = (lane >> 4) << 3;
    #pragma unroll
    for (int im = 0; im < 2; ++im) {
        #pragma unroll
        for (int in = 0; in < 2; ++in) {
            const int gn  = n_blk + wn * 32 + in * 16 + nl;
            const int gm0 = m_blk + wm * 32 + im * 16 + mh;
            const float bv = bias ? bias[gn] : 0.f;
            #pragma unroll
            for (int j = 0; j < 8; ++j) {
                float v = acc[im][in][j] + bv;
                if (act == ACT_RELU) v = fmaxf(v, 0.f);
                else if (act == ACT_GELU)
                    v = 0.5f * v * (1.f + erff(v * 0.70710678118654752f));
                if (res) v += (float)res[(size_t)(gm0 + j) * N + gn];
                D[(size_t)(gm0 + j) * N + gn] = (_Float16)v;
            }
        }
    }
}

// ---------------------------------------------------------------------------
// im2col for 3x3/pad1 convs, C=64; activations [H*W, 64] f16.
// col layout: [pix, tap*64 + c]  (K = 576)
// ---------------------------------------------------------------------------
__global__ void im2col3x3_c64(const _Float16* __restrict__ src,
                              _Float16* __restrict__ col, int H, int W)
{
    int idx = blockIdx.x * blockDim.x + threadIdx.x;
    int total = H * W * 9;
    if (idx >= total) return;
    int tap = idx % 9, pix = idx / 9;
    int y = pix / W, x = pix % W;
    int yy = y + tap / 3 - 1, xx = x + tap % 3 - 1;
    uint4* dst = (uint4*)(col + (size_t)pix * 576 + tap * 64);
    if (yy >= 0 && yy < H && xx >= 0 && xx < W) {
        const uint4* s = (const uint4*)(src + (size_t)(yy * W + xx) * 64);
        #pragma unroll
        for (int i = 0; i < 8; ++i) dst[i] = s[i];
    } else {
        uint4 z = {0u, 0u, 0u, 0u};
        #pragma unroll
        for (int i = 0; i < 8; ++i) dst[i] = z;
    }
}

// conv weight OIHW -> transposed GEMM weight Bt[(l,o)][tap*I+i], K = I*9
__global__ void conv_w_to_bt(const float* __restrict__ w,
                             _Float16* __restrict__ Bt, int L, int O, int I)
{
    int K = I * 9;
    long idx = (long)blockIdx.x * blockDim.x + threadIdx.x;
    long total = (long)L * O * K;
    if (idx >= total) return;
    int k = (int)(idx % K);
    long lo = idx / K;                 // l*O + o
    int tap = k / I, ii = k % I;
    Bt[idx] = (_Float16)w[(lo * I + ii) * 9 + tap];
}

// dense weight [N,K] f32 -> Bt [N][Kpad] f16 (zero pad K)
__global__ void dense_w_to_bt(const float* __restrict__ w,
                              _Float16* __restrict__ Bt, int N, int K, int Kpad)
{
    long idx = (long)blockIdx.x * blockDim.x + threadIdx.x;
    long total = (long)N * Kpad;
    if (idx >= total) return;
    int k = (int)(idx % Kpad);
    int n = (int)(idx / Kpad);
    Bt[idx] = (_Float16)((k < K) ? w[(long)n * K + k] : 0.f);
}

// EDSR head conv 1->64 (tiny, VALU); writes x and y copies
__global__ void head_conv1to64(const float* __restrict__ inp,
                               const float* __restrict__ w,
                               const float* __restrict__ b,
                               _Float16* __restrict__ x0,
                               _Float16* __restrict__ y0)
{
    int pix = blockIdx.x * blockDim.x + threadIdx.x;
    if (pix >= 128 * 128) return;
    int y = pix >> 7, x = pix & 127;
    float v[9];
    #pragma unroll
    for (int t = 0; t < 9; ++t) {
        int yy = y + t / 3 - 1, xx = x + t % 3 - 1;
        v[t] = (yy >= 0 && yy < 128 && xx >= 0 && xx < 128) ? inp[yy * 128 + xx] : 0.f;
    }
    for (int o = 0; o < 64; ++o) {
        float s = b[o];
        #pragma unroll
        for (int t = 0; t < 9; ++t) s += v[t] * w[o * 9 + t];
        _Float16 hv = (_Float16)s;
        x0[(size_t)pix * 64 + o] = hv;
        y0[(size_t)pix * 64 + o] = hv;
    }
}

// pixel shuffle r=2: u [16384,256] -> t [65536,64]
__global__ void pshuf2(const _Float16* __restrict__ u, _Float16* __restrict__ t)
{
    int idx = blockIdx.x * blockDim.x + threadIdx.x;
    if (idx >= 65536 * 64) return;
    int c = idx & 63, pix = idx >> 6;
    int oy = pix >> 8, ox = pix & 255;
    int h = oy >> 1, r1 = oy & 1, w = ox >> 1, r2 = ox & 1;
    t[idx] = u[(size_t)(h * 128 + w) * 256 + c * 4 + r1 * 2 + r2];
}

// tail conv 64->1 (tiny, VALU), vectorized f16 reads
__global__ void tail_conv64to1(const _Float16* __restrict__ t,
                               const float* __restrict__ w,
                               const float* __restrict__ b,
                               float* __restrict__ feat)
{
    int pix = blockIdx.x * blockDim.x + threadIdx.x;
    if (pix >= 256 * 256) return;
    int y = pix >> 8, x = pix & 255;
    float s = b[0];
    #pragma unroll
    for (int tp = 0; tp < 9; ++tp) {
        int yy = y + tp / 3 - 1, xx = x + tp % 3 - 1;
        if (yy >= 0 && yy < 256 && xx >= 0 && xx < 256) {
            const v16h* pv = (const v16h*)(t + (size_t)(yy * 256 + xx) * 64);
            #pragma unroll
            for (int g = 0; g < 4; ++g) {
                v16h q = pv[g];
                #pragma unroll
                for (int c = 0; c < 16; ++c)
                    s += (float)q[c] * w[(g * 16 + c) * 9 + tp];
            }
        }
    }
    feat[pix] = s;
}

// LIIF grid assembly: 14 channels (+pad to 32) per query pixel
__global__ void build_grid(const float* __restrict__ coord,
                           const float* __restrict__ cell,
                           const float* __restrict__ feat,
                           _Float16* __restrict__ grid)
{
    int pix = blockIdx.x * blockDim.x + threadIdx.x;
    if (pix >= 65536) return;
    float c0 = coord[(size_t)pix * 2 + 0];
    float c1 = coord[(size_t)pix * 2 + 1];
    const float Hf = 256.f, Wf = 256.f, rx = 1.f / 256.f, ry = 1.f / 256.f;
    float out[14], areas[4], fsv[4];
    int i = 0;
    #pragma unroll
    for (int vi = 0; vi < 2; ++vi) {
        #pragma unroll
        for (int vj = 0; vj < 2; ++vj) {
            float vx = vi ? 1.f : -1.f;
            float vy = vj ? 1.f : -1.f;
            float ch = fminf(fmaxf(c0 + vx * rx + 1e-6f, -1.f + 1e-6f), 1.f - 1e-6f);
            float cw = fminf(fmaxf(c1 + vy * ry + 1e-6f, -1.f + 1e-6f), 1.f - 1e-6f);
            int ih = (int)fminf(fmaxf(rintf((ch + 1.f) * Hf * 0.5f - 0.5f), 0.f), Hf - 1.f);
            int iw = (int)fminf(fmaxf(rintf((cw + 1.f) * Wf * 0.5f - 0.5f), 0.f), Wf - 1.f);
            fsv[i] = feat[ih * 256 + iw];
            float oh = -1.f + (2.f * ih + 1.f) / Hf;
            float ow = -1.f + (2.f * iw + 1.f) / Wf;
            float rh = (c0 - oh) * Hf, rw = (c1 - ow) * Wf;
            out[2 * i] = rh; out[2 * i + 1] = rw;
            areas[i] = fabsf(rh * rw) + 1e-9f;
            ++i;
        }
    }
    float tot = areas[0] + areas[1] + areas[2] + areas[3];
    #pragma unroll
    for (int j = 0; j < 4; ++j) out[8 + j] = fsv[j] * (areas[3 - j] / tot);
    out[12] = cell[0] * 256.f;
    out[13] = cell[1] * 256.f;
    _Float16* g = grid + (size_t)pix * 32;
    #pragma unroll
    for (int j = 0; j < 14; ++j) g[j] = (_Float16)out[j];
    #pragma unroll
    for (int j = 14; j < 32; ++j) g[j] = (_Float16)0.f;
}

// custom LN (Bessel var, eps on std) over k & v sections of qkv, in place
__global__ void ln_kv(_Float16* __restrict__ qkv,
                      const float* __restrict__ klnw, const float* __restrict__ klnb,
                      const float* __restrict__ vlnw, const float* __restrict__ vlnb,
                      int M)
{
    int idx = blockIdx.x * blockDim.x + threadIdx.x;
    if (idx >= M * 32) return;
    int part = idx & 1;            // 0=k, 1=v
    int h = (idx >> 1) & 15;
    int n = idx >> 5;
    _Float16* p = qkv + (size_t)n * 768 + h * 48 + 16 + part * 16;  // 32B aligned
    v16h xh = *(const v16h*)p;
    float xv[16], mean = 0.f;
    #pragma unroll
    for (int j = 0; j < 16; ++j) { xv[j] = (float)xh[j]; mean += xv[j]; }
    mean *= (1.f / 16.f);
    float var = 0.f;
    #pragma unroll
    for (int j = 0; j < 16; ++j) { float d = xv[j] - mean; var += d * d; }
    var *= (1.f / 15.f);
    float inv = 1.f / (sqrtf(var) + 1e-5f);
    const float* wv = part ? vlnw : klnw;
    const float* bv = part ? vlnb : klnb;
    v16h yh;
    #pragma unroll
    for (int j = 0; j < 16; ++j)
        yh[j] = (_Float16)(wv[h * 16 + j] * ((xv[j] - mean) * inv) + bv[h * 16 + j]);
    *(v16h*)p = yh;
}

__global__ void fill_zero_f32(float* p, int n)
{
    int i = blockIdx.x * blockDim.x + threadIdx.x;
    if (i < n) p[i] = 0.f;
}

// kv[h,c,d] = sum_n k[n,h,c] * v[n,h,d] / M  via WMMA (A = k^T, B = v).
// 32-pixel tiles staged transposed ([channel][pixel]) so both fragments are
// contiguous ds_load_b128 runs.
__global__ __launch_bounds__(256) void attn_kv_wmma(
    const _Float16* __restrict__ qkv, float* __restrict__ kv, int M)
{
    __shared__ alignas(16) _Float16 st[8][32][40];  // [wave][ch: k0-15|v16-31][pix]
    const int lane = threadIdx.x & 31;
    const int wave = threadIdx.x >> 5;
    const int h = blockIdx.x;
    const int pixPerBlock = M / gridDim.y;
    const int pixPerWave = pixPerBlock / 8;
    const int base = blockIdx.y * pixPerBlock + wave * pixPerWave;

    v8f acc = {};
    for (int p0 = 0; p0 < pixPerWave; p0 += 32) {
        const _Float16* src = qkv + (size_t)(base + p0 + lane) * 768 + h * 48 + 16;
        v16h kvals = *(const v16h*)src;         // 32B aligned
        v16h vvals = *(const v16h*)(src + 16);
        #pragma unroll
        for (int c = 0; c < 16; ++c) {          // transpose scatter (stores only)
            st[wave][c][lane]      = kvals[c];
            st[wave][16 + c][lane] = vvals[c];
        }
        __syncthreads();

        const int rc = lane & 15;
        const int ka = (lane >> 4) << 3;
        const int kb = (lane >> 4) << 3;
        const _Float16* ap = &st[wave][rc][ka];        // A[c][n] = k^T
        const _Float16* bp = &st[wave][16 + rc][kb];   // B[n][d] = v (col rc)
        v16h af, bf;
        #pragma unroll
        for (int i = 0; i < 8; ++i) {
            int k = (i < 4) ? (i << 1) : (16 + ((i - 4) << 1));
            af[2 * i]     = ap[k];
            af[2 * i + 1] = ap[k + 1];
            bf[2 * i]     = bp[k];
            bf[2 * i + 1] = bp[k + 1];
        }
        acc = __builtin_amdgcn_wmma_f32_16x16x32_f16(
            false, af, false, bf, (short)0, acc, false, false);
        __syncthreads();
    }
    const int d = lane & 15;
    const int c0 = (lane >> 4) << 3;
    const float scale = 1.f / (float)M;
    #pragma unroll
    for (int j = 0; j < 8; ++j)
        atomicAdd(&kv[h * 256 + (c0 + j) * 16 + d], acc[j] * scale);
}

// ret[n, h*16+d] = sum_c q[n,h,c] * kv[h,c,d] + x[n, h*16+d]
__global__ __launch_bounds__(256) void attn_o_add(
    const _Float16* __restrict__ qkv, const float* __restrict__ kv,
    const _Float16* __restrict__ xin, _Float16* __restrict__ ret, int M)
{
    __shared__ float kvs[4096];
    for (int i = threadIdx.x; i < 4096; i += blockDim.x) kvs[i] = kv[i];
    __syncthreads();
    int n = blockIdx.x * blockDim.x + threadIdx.x;
    if (n >= M) return;
    const _Float16* qp = qkv + (size_t)n * 768;
    const v16h* xp = (const v16h*)(xin + (size_t)n * 256);
    v16h* rp = (v16h*)(ret + (size_t)n * 256);
    for (int hh = 0; hh < 16; ++hh) {
        v16h qh = *(const v16h*)(qp + hh * 48);   // 32B aligned (96*h mult of 32)
        float q[16];
        #pragma unroll
        for (int c = 0; c < 16; ++c) q[c] = (float)qh[c];
        v16h xv = xp[hh];
        v16h ov;
        #pragma unroll
        for (int d = 0; d < 16; ++d) {
            float s = 0.f;
            #pragma unroll
            for (int c = 0; c < 16; ++c) s += q[c] * kvs[hh * 256 + c * 16 + d];
            ov[d] = (_Float16)(s + (float)xv[d]);
        }
        rp[hh] = ov;
    }
}

// fc2 (256->1) + bilinear-border skip from the LR input
__global__ void final_out(const _Float16* __restrict__ f,
                          const float* __restrict__ w2, const float* __restrict__ b2,
                          const float* __restrict__ inp,
                          const float* __restrict__ coord,
                          float* __restrict__ out)
{
    int pix = blockIdx.x * blockDim.x + threadIdx.x;
    if (pix >= 65536) return;
    float s = b2[0];
    const v16h* fp = (const v16h*)(f + (size_t)pix * 256);
    #pragma unroll
    for (int g = 0; g < 16; ++g) {
        v16h q = fp[g];
        #pragma unroll
        for (int c = 0; c < 16; ++c) s += (float)q[c] * w2[g * 16 + c];
    }
    float c0 = coord[(size_t)pix * 2 + 0];
    float c1 = coord[(size_t)pix * 2 + 1];
    float y = fminf(fmaxf((c0 + 1.f) * 64.f - 0.5f, 0.f), 127.f);
    float x = fminf(fmaxf((c1 + 1.f) * 64.f - 0.5f, 0.f), 127.f);
    float y0f = floorf(y), x0f = floorf(x);
    int y0 = (int)y0f, x0 = (int)x0f;
    int y1 = min(y0 + 1, 127), x1 = min(x0 + 1, 127);
    float wy = y - y0f, wx = x - x0f;
    float v00 = inp[y0 * 128 + x0], v01 = inp[y0 * 128 + x1];
    float v10 = inp[y1 * 128 + x0], v11 = inp[y1 * 128 + x1];
    s += v00 * (1.f - wy) * (1.f - wx) + v01 * (1.f - wy) * wx +
         v10 * wy * (1.f - wx) + v11 * wy * wx;
    out[pix] = s;
}

// ---------------------------------------------------------------------------
// Host side
// ---------------------------------------------------------------------------
static inline void launch_gemm(const _Float16* A, const _Float16* Bt,
                               const float* bias, const _Float16* res,
                               _Float16* D, int M, int N, int K, int act,
                               hipStream_t stream)
{
    dim3 g(M / BM, N / BN);
    gemm_f16_wmma<<<g, 256, 0, stream>>>(A, Bt, bias, res, D, M, N, K, act);
}

extern "C" void kernel_launch(void* const* d_in, const int* in_sizes, int n_in,
                              void* d_out, int out_size, void* d_ws, size_t ws_size,
                              hipStream_t stream)
{
    (void)in_sizes; (void)n_in; (void)out_size; (void)ws_size;
    const float* inp    = (const float*)d_in[0];
    const float* coord  = (const float*)d_in[1];
    const float* cell   = (const float*)d_in[2];
    const float* head_w = (const float*)d_in[3];
    const float* head_b = (const float*)d_in[4];
    const float* res_w1 = (const float*)d_in[5];
    const float* res_b1 = (const float*)d_in[6];
    const float* res_w2 = (const float*)d_in[7];
    const float* res_b2 = (const float*)d_in[8];
    const float* bt_w   = (const float*)d_in[9];
    const float* bt_b   = (const float*)d_in[10];
    const float* up_w   = (const float*)d_in[11];
    const float* up_b   = (const float*)d_in[12];
    const float* tail_w = (const float*)d_in[13];
    const float* tail_b = (const float*)d_in[14];
    const float* c00_w  = (const float*)d_in[15];
    const float* c00_b  = (const float*)d_in[16];
    const float* fc1_w  = (const float*)d_in[17];
    const float* fc1_b  = (const float*)d_in[18];
    const float* fc2_w  = (const float*)d_in[19];
    const float* fc2_b  = (const float*)d_in[20];

    struct AttnP {
        const float *qkv_w, *qkv_b, *o1_w, *o1_b, *o2_w, *o2_b;
        const float *kln_w, *kln_b, *vln_w, *vln_b;
    };
    AttnP at[2];
    for (int a = 0; a < 2; ++a) {
        int b0 = 21 + a * 10;
        at[a].qkv_w = (const float*)d_in[b0 + 0];
        at[a].qkv_b = (const float*)d_in[b0 + 1];
        at[a].o1_w  = (const float*)d_in[b0 + 2];
        at[a].o1_b  = (const float*)d_in[b0 + 3];
        at[a].o2_w  = (const float*)d_in[b0 + 4];
        at[a].o2_b  = (const float*)d_in[b0 + 5];
        at[a].kln_w = (const float*)d_in[b0 + 6];
        at[a].kln_b = (const float*)d_in[b0 + 7];
        at[a].vln_w = (const float*)d_in[b0 + 8];
        at[a].vln_b = (const float*)d_in[b0 + 9];
    }
    float* out = (float*)d_out;

    // ---- workspace carve-up (bump allocator, 256B aligned) ----
    char* wp = (char*)d_ws;
    auto alloc = [&](size_t bytes) -> char* {
        char* p = wp;
        wp += (bytes + 255) & ~(size_t)255;
        return p;
    };
    const int M_LR = 128 * 128;   // 16384
    const int M_HR = 256 * 256;   // 65536

    _Float16* colb  = (_Float16*)alloc((size_t)M_LR * 576 * 2);
    _Float16* xbuf  = (_Float16*)alloc((size_t)M_LR * 64 * 2);
    _Float16* ybufA = (_Float16*)alloc((size_t)M_LR * 64 * 2);
    _Float16* ybufB = (_Float16*)alloc((size_t)M_LR * 64 * 2);
    _Float16* hbufE = (_Float16*)alloc((size_t)M_LR * 64 * 2);
    _Float16* ubuf  = (_Float16*)alloc((size_t)M_LR * 256 * 2);
    _Float16* tbuf  = (_Float16*)alloc((size_t)M_HR * 64 * 2);
    float*    featb = (float*)   alloc((size_t)M_HR * 4);
    _Float16* gridb = (_Float16*)alloc((size_t)M_HR * 32 * 2);
    _Float16* xA    = (_Float16*)alloc((size_t)M_HR * 256 * 2);
    _Float16* xB    = (_Float16*)alloc((size_t)M_HR * 256 * 2);
    _Float16* retb  = (_Float16*)alloc((size_t)M_HR * 256 * 2);
    _Float16* h2b   = (_Float16*)alloc((size_t)M_HR * 256 * 2);
    _Float16* qkvb  = (_Float16*)alloc((size_t)M_HR * 768 * 2);
    float*    kvb   = (float*)   alloc(4096 * 4);
    // transposed f16 GEMM weights (Bt layout [N][Kpad])
    _Float16* w1B   = (_Float16*)alloc((size_t)16 * 64 * 576 * 2);
    _Float16* w2B   = (_Float16*)alloc((size_t)16 * 64 * 576 * 2);
    _Float16* btB   = (_Float16*)alloc((size_t)64 * 576 * 2);
    _Float16* upB   = (_Float16*)alloc((size_t)256 * 576 * 2);
    _Float16* c00B  = (_Float16*)alloc((size_t)256 * 32 * 2);
    _Float16* qkvB[2] = { (_Float16*)alloc((size_t)768 * 256 * 2),
                          (_Float16*)alloc((size_t)768 * 256 * 2) };
    _Float16* o1B[2]  = { (_Float16*)alloc((size_t)256 * 256 * 2),
                          (_Float16*)alloc((size_t)256 * 256 * 2) };
    _Float16* o2B[2]  = { (_Float16*)alloc((size_t)256 * 256 * 2),
                          (_Float16*)alloc((size_t)256 * 256 * 2) };
    _Float16* fc1B    = (_Float16*)alloc((size_t)256 * 256 * 2);

    // ---- weight conversion (every call; deterministic) ----
    {
        long t;
        t = 16L * 64 * 576;
        conv_w_to_bt<<<dim3((t + 255) / 256), 256, 0, stream>>>(res_w1, w1B, 16, 64, 64);
        conv_w_to_bt<<<dim3((t + 255) / 256), 256, 0, stream>>>(res_w2, w2B, 16, 64, 64);
        t = 64L * 576;
        conv_w_to_bt<<<dim3((t + 255) / 256), 256, 0, stream>>>(bt_w, btB, 1, 64, 64);
        t = 256L * 576;
        conv_w_to_bt<<<dim3((t + 255) / 256), 256, 0, stream>>>(up_w, upB, 1, 256, 64);
        t = 256L * 32;
        dense_w_to_bt<<<dim3((t + 255) / 256), 256, 0, stream>>>(c00_w, c00B, 256, 14, 32);
        t = 768L * 256;
        for (int a = 0; a < 2; ++a)
            dense_w_to_bt<<<dim3((t + 255) / 256), 256, 0, stream>>>(at[a].qkv_w, qkvB[a], 768, 256, 256);
        t = 256L * 256;
        for (int a = 0; a < 2; ++a) {
            dense_w_to_bt<<<dim3((t + 255) / 256), 256, 0, stream>>>(at[a].o1_w, o1B[a], 256, 256, 256);
            dense_w_to_bt<<<dim3((t + 255) / 256), 256, 0, stream>>>(at[a].o2_w, o2B[a], 256, 256, 256);
        }
        dense_w_to_bt<<<dim3((t + 255) / 256), 256, 0, stream>>>(fc1_w, fc1B, 256, 256, 256);
    }

    // ---- EDSR backbone ----
    head_conv1to64<<<dim3((M_LR + 255) / 256), 256, 0, stream>>>(inp, head_w, head_b, xbuf, ybufA);
    _Float16* ycur = ybufA;
    _Float16* ynxt = ybufB;
    const int IMG = (M_LR * 9 + 255) / 256;
    for (int i = 0; i < 16; ++i) {
        im2col3x3_c64<<<dim3(IMG), 256, 0, stream>>>(ycur, colb, 128, 128);
        launch_gemm(colb, w1B + (size_t)i * 64 * 576, res_b1 + i * 64, nullptr,
                    hbufE, M_LR, 64, 576, ACT_RELU, stream);
        im2col3x3_c64<<<dim3(IMG), 256, 0, stream>>>(hbufE, colb, 128, 128);
        launch_gemm(colb, w2B + (size_t)i * 64 * 576, res_b2 + i * 64, ycur,
                    ynxt, M_LR, 64, 576, ACT_NONE, stream);
        _Float16* tmp = ycur; ycur = ynxt; ynxt = tmp;
    }
    im2col3x3_c64<<<dim3(IMG), 256, 0, stream>>>(ycur, colb, 128, 128);
    launch_gemm(colb, btB, bt_b, xbuf, ynxt, M_LR, 64, 576, ACT_NONE, stream);
    { _Float16* tmp = ycur; ycur = ynxt; ynxt = tmp; }
    im2col3x3_c64<<<dim3(IMG), 256, 0, stream>>>(ycur, colb, 128, 128);
    launch_gemm(colb, upB, up_b, nullptr, ubuf, M_LR, 256, 576, ACT_NONE, stream);
    pshuf2<<<dim3((M_HR * 64 + 255) / 256), 256, 0, stream>>>(ubuf, tbuf);
    tail_conv64to1<<<dim3((M_HR + 255) / 256), 256, 0, stream>>>(tbuf, tail_w, tail_b, featb);

    // ---- query head ----
    build_grid<<<dim3((M_HR + 255) / 256), 256, 0, stream>>>(coord, cell, featb, gridb);
    launch_gemm(gridb, c00B, c00_b, nullptr, xA, M_HR, 256, 32, ACT_NONE, stream);

    _Float16* xin = xA;
    _Float16* xout = xB;
    for (int a = 0; a < 2; ++a) {
        launch_gemm(xin, qkvB[a], at[a].qkv_b, nullptr, qkvb, M_HR, 768, 256, ACT_NONE, stream);
        ln_kv<<<dim3((M_HR * 32 + 255) / 256), 256, 0, stream>>>(
            qkvb, at[a].kln_w, at[a].kln_b, at[a].vln_w, at[a].vln_b, M_HR);
        fill_zero_f32<<<dim3(16), 256, 0, stream>>>(kvb, 4096);
        attn_kv_wmma<<<dim3(16, 32), 256, 0, stream>>>(qkvb, kvb, M_HR);
        attn_o_add<<<dim3((M_HR + 255) / 256), 256, 0, stream>>>(qkvb, kvb, xin, retb, M_HR);
        launch_gemm(retb, o1B[a], at[a].o1_b, nullptr, h2b, M_HR, 256, 256, ACT_GELU, stream);
        launch_gemm(h2b, o2B[a], at[a].o2_b, /*res=*/xin, xout, M_HR, 256, 256, ACT_NONE, stream);
        _Float16* tmp = xin; xin = xout; xout = tmp;
    }
    launch_gemm(xin, fc1B, fc1_b, nullptr, h2b, M_HR, 256, 256, ACT_GELU, stream);
    final_out<<<dim3((M_HR + 255) / 256), 256, 0, stream>>>(h2b, fc2_w, fc2_b, inp, coord, out);
}
